// Spares3dELAN_55387898249947
// MI455X (gfx1250) — matrix operand order repaired and checked by reading, not compiled, not run on hardware
//
#include <hip/hip_runtime.h>
#include <hip/hip_bf16.h>

// MI455X / gfx1250 implementation.
// Workspace layout: cat_h = N x 256 f16 activation/concat buffer (134 MB for
// N=262144; fits comfortably in the 192 MB L2, so the 27-tap gathers are
// L2-resident). All GEMMs use v_wmma_f32_16x16x32_f16 with f32 accumulation;
// LayerNorm runs on the f32 accumulators.

typedef __attribute__((ext_vector_type(8)))  _Float16 v8h;
typedef __attribute__((ext_vector_type(16))) _Float16 v16h;
typedef __attribute__((ext_vector_type(8)))  float    v8f;

#define LN_EPS 1e-5f

// ---- WMMA fragment helpers (layouts per CDNA5 ISA 7.12.2, wave32) ----------

// A (16x32 f16): lane<16 -> row=lane, K in {hi8+0..7} U {16+hi8+0..7} with
// hi8 = (lane>=16)*8.  Two contiguous 16B chunks per lane.
__device__ __forceinline__ v16h a_frag_from_row(const _Float16* rowp, int hi8) {
  v8h c0 = *(const v8h*)(rowp + hi8);
  v8h c1 = *(const v8h*)(rowp + 16 + hi8);
  v16h a;
#pragma unroll
  for (int e = 0; e < 8; ++e) { a[e] = c0[e]; a[e + 8] = c1[e]; }
  return a;
}

// B (32x16 f16) pre-packed into LDS so each lane reads 16 contiguous halfs.
__device__ __forceinline__ v16h b_frag(const _Float16* packed, int lane) {
  return *(const v16h*)(packed + lane * 16);
}

// Convert f32 weight [KD][NC] (row-major) into fragment-ordered f16 LDS:
// region (kkTile, nTile) of 512 halfs, inner layout [lane][e] so b_frag() is
// a single contiguous 32B ds read.
template <int KD, int NC>
__device__ __forceinline__ void pack_weight_f16(const float* __restrict__ src,
                                                _Float16* dst, int tid, int nth) {
  constexpr int NT = NC / 16;
  for (int idx = tid; idx < KD * NC; idx += nth) {
    int kd = idx / NC, n = idx - kd * NC;
    int kkt = kd >> 5, rem = kd & 31, hi = (rem >> 3) & 1, e = (rem & 7) | ((rem >> 4) << 3);
    int nt = n >> 4, lane = (hi << 4) | (n & 15);
    dst[((kkt * NT + nt) << 9) + lane * 16 + e] = (_Float16)src[idx];
  }
}

__device__ __forceinline__ v8f wmma_f16(v16h a, v16h b, v8f c) {
  return __builtin_amdgcn_wmma_f32_16x16x32_f16(false, a, false, b, (short)0, c,
                                                false, false);
}

// ---- Kernel 1: f1 = blk1(x,W1), f2 = blk1(x,W2) -> cat[:,0:64], cat[:,64:128]
__global__ __launch_bounds__(128)
void k_dual_blk1(const float* __restrict__ x,
                 const float* __restrict__ W1, const float* __restrict__ b1,
                 const float* __restrict__ g1, const float* __restrict__ be1,
                 const float* __restrict__ W2, const float* __restrict__ b2,
                 const float* __restrict__ g2, const float* __restrict__ be2,
                 _Float16* __restrict__ cat, int N) {
  __shared__ _Float16 xh[64 * 64];      //  8 KB activations (f16)
  __shared__ _Float16 wp[2][64 * 64];   // 16 KB packed W1,W2
  __shared__ float outl[64 * 64];       // 16 KB f32 pre-LN tile

  const int tid = threadIdx.x;
  const int lane = tid & 31, wave = tid >> 5;
  const int rowBase = blockIdx.x * 64;

  for (int i = tid; i < 64 * 64; i += 128) {
    int r = i >> 6;
    int gr = rowBase + r;
    int grc = gr < N ? gr : N - 1;
    xh[i] = (_Float16)x[(long)grc * 64 + (i & 63)];
  }
  pack_weight_f16<64, 64>(W1, wp[0], tid, 128);
  pack_weight_f16<64, 64>(W2, wp[1], tid, 128);
  __syncthreads();

  const int hi = lane >> 4, hi8 = hi << 3;
  const int m = lane & 15;
  const int lr = (wave << 4) | m;
  const _Float16* rowp = xh + lr * 64;
  v16h a0 = a_frag_from_row(rowp, hi8);
  v16h a1 = a_frag_from_row(rowp + 32, hi8);

  const float* Bv[2] = {b1, b2};
  const float* Gv[2] = {g1, g2};
  const float* BEv[2] = {be1, be2};

  for (int pass = 0; pass < 2; ++pass) {
    if (pass) __syncthreads();          // prev LN reads of outl finished
#pragma unroll
    for (int nt = 0; nt < 4; ++nt) {
      v8f acc = {};
      acc = wmma_f16(a0, b_frag(wp[pass] + ((0 * 4 + nt) << 9), lane), acc);
      acc = wmma_f16(a1, b_frag(wp[pass] + ((1 * 4 + nt) << 9), lane), acc);
      int n = (nt << 4) | m;
      float bn = Bv[pass][n];
#pragma unroll
      for (int r = 0; r < 8; ++r)
        outl[((wave << 4) + (hi << 3) + r) * 64 + n] = acc[r] + bn;
    }
    __syncthreads();
    if (tid < 64) {                      // one thread per row: LN + ReLU
      int gr = rowBase + tid;
      float s = 0.f, ss = 0.f;
      for (int c = 0; c < 64; ++c) {
        float v = outl[tid * 64 + c]; s += v; ss += v * v;
      }
      float mu = s * (1.f / 64.f);
      float inv = rsqrtf(ss * (1.f / 64.f) - mu * mu + LN_EPS);
      if (gr < N) {
        _Float16* dst = cat + (long)gr * 256 + pass * 64;
        for (int c = 0; c < 64; ++c) {
          float y = (outl[tid * 64 + c] - mu) * inv * Gv[pass][c] + BEv[pass][c];
          dst[c] = (_Float16)fmaxf(y, 0.f);
        }
      }
    }
  }
}

// ---- Submanifold conv block: out = relu(ln(b + sum_k gather(x,nbr_k) @ W_k))
// Reads cat[:, inOff:inOff+INW], writes cat[:, outOff:outOff+32].
template <int INW>
__global__ __launch_bounds__(128)
void k_subm_conv(const int* __restrict__ nbr, const float* __restrict__ W,
                 const float* __restrict__ b, const float* __restrict__ g,
                 const float* __restrict__ be, _Float16* __restrict__ cat,
                 int inOff, int outOff, int N) {
  __shared__ _Float16 wbuf[2][INW * 32];  // double-buffered per-tap weights
  __shared__ float outl[64 * 32];

  const int tid = threadIdx.x, lane = tid & 31, wave = tid >> 5;
  const int rowBase = blockIdx.x * 64;
  const int hi = lane >> 4, hi8 = hi << 3;
  const int m = lane & 15;
  const int gr = rowBase + (wave << 4) + m;
  const int grc = gr < N ? gr : N - 1;
  constexpr int NKK = INW / 32;

  v8f acc[2] = {{}, {}};

  pack_weight_f16<INW, 32>(W, wbuf[0], tid, 128);  // tap 0

  for (int k = 0; k < 27; ++k) {
    __syncthreads();                                // slab k ready; k-1 consumed
    if (k + 1 < 27)
      pack_weight_f16<INW, 32>(W + (long)(k + 1) * INW * 32, wbuf[(k + 1) & 1],
                               tid, 128);
    int idx = nbr[(long)grc * 27 + k];
    const _Float16* rowp = cat + (long)(idx < 0 ? 0 : idx) * 256 + inOff;
    const _Float16* wk = wbuf[k & 1];
    if (k + 1 < 27) {                               // prefetch next gather row
      int idn = nbr[(long)grc * 27 + k + 1];
      if (idn >= 0) __builtin_prefetch(cat + (long)idn * 256 + inOff, 0, 1);
    }
#pragma unroll
    for (int kk = 0; kk < NKK; ++kk) {
      v16h a = a_frag_from_row(rowp + kk * 32, hi8);
      if (idx < 0) { v16h z = {}; a = z; }          // masked taps contribute 0
      acc[0] = wmma_f16(a, b_frag(wk + ((kk * 2 + 0) << 9), lane), acc[0]);
      acc[1] = wmma_f16(a, b_frag(wk + ((kk * 2 + 1) << 9), lane), acc[1]);
    }
  }

#pragma unroll
  for (int nt = 0; nt < 2; ++nt) {
    int n = (nt << 4) | m;
    float bn = b[n];
#pragma unroll
    for (int r = 0; r < 8; ++r)
      outl[((wave << 4) + (hi << 3) + r) * 32 + n] = acc[nt][r] + bn;
  }
  __syncthreads();
  if (tid < 64) {
    int grr = rowBase + tid;
    float s = 0.f, ss = 0.f;
    for (int c = 0; c < 32; ++c) { float v = outl[tid * 32 + c]; s += v; ss += v * v; }
    float mu = s * (1.f / 32.f);
    float inv = rsqrtf(ss * (1.f / 32.f) - mu * mu + LN_EPS);
    if (grr < N) {
      _Float16* dst = cat + (long)grr * 256 + outOff;
      for (int c = 0; c < 32; ++c) {
        float y = (outl[tid * 32 + c] - mu) * inv * g[c] + be[c];
        dst[c] = (_Float16)fmaxf(y, 0.f);
      }
    }
  }
}

// ---- Final: out = relu(ln(cat @ W7 + b7)) , cat is N x 256 f16 ------------
__global__ __launch_bounds__(128)
void k_final(const _Float16* __restrict__ cat, const float* __restrict__ W7,
             const float* __restrict__ b7, const float* __restrict__ g7,
             const float* __restrict__ be7, float* __restrict__ out, int N) {
  __shared__ _Float16 wp[256 * 64];  // 32 KB packed W7
  __shared__ float outl[64 * 64];    // 16 KB

  const int tid = threadIdx.x, lane = tid & 31, wave = tid >> 5;
  const int rowBase = blockIdx.x * 64;
  const int hi = lane >> 4, hi8 = hi << 3;
  const int m = lane & 15;
  const int gr = rowBase + (wave << 4) + m;
  const int grc = gr < N ? gr : N - 1;

  pack_weight_f16<256, 64>(W7, wp, tid, 128);
  __syncthreads();

  const _Float16* rowp = cat + (long)grc * 256;
  v8f acc[4] = {{}, {}, {}, {}};
#pragma unroll
  for (int kt = 0; kt < 8; ++kt) {
    v16h a = a_frag_from_row(rowp + kt * 32, hi8);
#pragma unroll
    for (int nt = 0; nt < 4; ++nt)
      acc[nt] = wmma_f16(a, b_frag(wp + ((kt * 4 + nt) << 9), lane), acc[nt]);
  }
#pragma unroll
  for (int nt = 0; nt < 4; ++nt) {
    int n = (nt << 4) | m;
    float bn = b7[n];
#pragma unroll
    for (int r = 0; r < 8; ++r)
      outl[((wave << 4) + (hi << 3) + r) * 64 + n] = acc[nt][r] + bn;
  }
  __syncthreads();
  if (tid < 64) {
    int grr = rowBase + tid;
    float s = 0.f, ss = 0.f;
    for (int c = 0; c < 64; ++c) { float v = outl[tid * 64 + c]; s += v; ss += v * v; }
    float mu = s * (1.f / 64.f);
    float inv = rsqrtf(ss * (1.f / 64.f) - mu * mu + LN_EPS);
    if (grr < N) {
      for (int c = 0; c < 64; ++c) {
        float y = (outl[tid * 64 + c] - mu) * inv * g7[c] + be7[c];
        out[(long)grr * 64 + c] = fmaxf(y, 0.f);
      }
    }
  }
}

extern "C" void kernel_launch(void* const* d_in, const int* in_sizes, int n_in,
                              void* d_out, int out_size, void* d_ws, size_t ws_size,
                              hipStream_t stream) {
  const float* x   = (const float*)d_in[0];
  const int*   nbr = (const int*)d_in[1];
  const float* W1 = (const float*)d_in[2],  *b1 = (const float*)d_in[3];
  const float* g1 = (const float*)d_in[4],  *be1 = (const float*)d_in[5];
  const float* W2 = (const float*)d_in[6],  *b2 = (const float*)d_in[7];
  const float* g2 = (const float*)d_in[8],  *be2 = (const float*)d_in[9];
  const float* W3 = (const float*)d_in[10], *b3 = (const float*)d_in[11];
  const float* g3 = (const float*)d_in[12], *be3 = (const float*)d_in[13];
  const float* W4 = (const float*)d_in[14], *b4 = (const float*)d_in[15];
  const float* g4 = (const float*)d_in[16], *be4 = (const float*)d_in[17];
  const float* W5 = (const float*)d_in[18], *b5 = (const float*)d_in[19];
  const float* g5 = (const float*)d_in[20], *be5 = (const float*)d_in[21];
  const float* W6 = (const float*)d_in[22], *b6 = (const float*)d_in[23];
  const float* g6 = (const float*)d_in[24], *be6 = (const float*)d_in[25];
  const float* W7 = (const float*)d_in[26], *b7 = (const float*)d_in[27];
  const float* g7 = (const float*)d_in[28], *be7 = (const float*)d_in[29];

  const int N = in_sizes[0] / 64;            // 262144
  _Float16* cat = (_Float16*)d_ws;           // N x 256 f16 (134 MB; L2-resident)

  const int nb = (N + 63) / 64;
  dim3 blk(128);                              // 4 wave32 waves; 16 rows each
  k_dual_blk1<<<nb, blk, 0, stream>>>(x, W1, b1, g1, be1, W2, b2, g2, be2, cat, N);
  k_subm_conv<64><<<nb, blk, 0, stream>>>(nbr, W3, b3, g3, be3, cat, 64, 128, N);
  k_subm_conv<32><<<nb, blk, 0, stream>>>(nbr, W4, b4, g4, be4, cat, 128, 160, N);
  k_subm_conv<32><<<nb, blk, 0, stream>>>(nbr, W5, b5, g5, be5, cat, 160, 192, N);
  k_subm_conv<32><<<nb, blk, 0, stream>>>(nbr, W6, b6, g6, be6, cat, 192, 224, N);
  k_final<<<nb, blk, 0, stream>>>(cat, W7, b7, g7, be7, (float*)d_out, N);
}
